// MultiHeadAttention_65867618451855
// MI455X (gfx1250) — matrix operand order repaired and checked by reading, not compiled, not run on hardware
//
#include <hip/hip_runtime.h>

// ---------------------------------------------------------------------------
// MHA forward for MI455X (gfx1250, wave32, WMMA, 320KB LDS).
// B=2, S=2048, D_MODEL=1024, H=16, Dh=64.
// d_out = [out f32 4,194,304][attn f32 134,217,728]
// Pipeline: cast -> Q/K/V proj (WMMA bf16) -> fused scores+softmax+AV
//           (scores strip lives in LDS) -> output proj (WMMA bf16).
// ---------------------------------------------------------------------------

typedef __attribute__((ext_vector_type(16))) __bf16 v16bf;
typedef __attribute__((ext_vector_type(8)))  __bf16 v8bf;
typedef __attribute__((ext_vector_type(4)))  __bf16 v4bf;
typedef __attribute__((ext_vector_type(8)))  float  v8f;
typedef __attribute__((ext_vector_type(4)))  float  v4f;

union V16U { v16bf v; v8bf h[2]; };
union V8U  { v8bf v; v4bf h[2]; };

__device__ __forceinline__ v8f zero8() {
    v8f z = {0.f,0.f,0.f,0.f,0.f,0.f,0.f,0.f};
    return z;
}

// A fragment (16x32 bf16, M x K), ISA wave32 layout:
//  lanes 0-15 : K = [kb+0..7]  and [kb+16..23]
//  lanes 16-31: K = [kb+8..15] and [kb+24..31]
__device__ __forceinline__ v16bf load_a_bf16(const __bf16* __restrict__ row, int kb, int half) {
    V16U u;
    u.h[0] = *(const v8bf*)(row + kb + 8 * half);
    u.h[1] = *(const v8bf*)(row + kb + 16 + 8 * half);
    return u.v;
}

// Same A layout, source f32 in LDS (attention probabilities), convert to bf16.
__device__ __forceinline__ v16bf load_a_ldsf32(const float* row, int kb, int half) {
    v4f f0 = *(const v4f*)(row + kb + 8 * half);
    v4f f1 = *(const v4f*)(row + kb + 8 * half + 4);
    v4f f2 = *(const v4f*)(row + kb + 16 + 8 * half);
    v4f f3 = *(const v4f*)(row + kb + 16 + 8 * half + 4);
    V8U lo, hi;
    lo.h[0] = __builtin_convertvector(f0, v4bf);
    lo.h[1] = __builtin_convertvector(f1, v4bf);
    hi.h[0] = __builtin_convertvector(f2, v4bf);
    hi.h[1] = __builtin_convertvector(f3, v4bf);
    V16U u; u.h[0] = lo.v; u.h[1] = hi.v;
    return u.v;
}

// B fragment (32x16 bf16, K x N): lane n = lane&15 holds B[kb+8*half+i][n] and B[kb+16+8*half+i][n], i=0..7.
// With B stored as [N][K] row-major this is two runs of 8 contiguous bf16 of row n, starting 16 apart.
__device__ __forceinline__ v16bf load_b_bf16(const __bf16* __restrict__ rowN, int kb, int half) {
    const __bf16* p = rowN + kb + 8 * half;
    V16U u;
    u.h[0] = *(const v8bf*)(p);
    u.h[1] = *(const v8bf*)(p + 16);
    return u.v;
}

__device__ __forceinline__ v8f wmma_bf16(v16bf a, v16bf b, v8f c) {
    return __builtin_amdgcn_wmma_f32_16x16x32_bf16(false, a, false, b, (short)0, c, false, false);
}

// ---------------------------------------------------------------------------
// f32 -> bf16 cast
// ---------------------------------------------------------------------------
__global__ __launch_bounds__(256) void cast_f32_bf16(const float* __restrict__ in,
                                                     __bf16* __restrict__ out, int n) {
    int i = blockIdx.x * 256 + threadIdx.x;
    if (i < n) out[i] = (__bf16)in[i];
}

// ---------------------------------------------------------------------------
// QKV projection: C[4096,1024] = X x W^T + bias, head-major routing.
// Wave tile 16(M) x 64(N), K=1024, register double-buffered loads.
// mode 0/1: out[b,h,s,dh]   mode 2: out[b,h,dh,s] (V transposed for AV)
// ---------------------------------------------------------------------------
__global__ __launch_bounds__(256) void proj_qkv_kernel(const __bf16* __restrict__ X,
                                                       const __bf16* __restrict__ W,
                                                       const float* __restrict__ bias,
                                                       __bf16* __restrict__ out, int mode) {
    const int lane = threadIdx.x & 31;
    const int wid  = blockIdx.x * 8 + (threadIdx.x >> 5);
    const int strip = wid & 15;   // 1024/64 N strips
    const int mtile = wid >> 4;   // 4096/16 M tiles
    const int m0 = mtile * 16, n0 = strip * 64;
    const int half = lane >> 4, l16 = lane & 15;

    const __bf16* arow = X + (long)(m0 + l16) * 1024;
    const __bf16* b0r = W + (long)(n0 +  0 + l16) * 1024;
    const __bf16* b1r = W + (long)(n0 + 16 + l16) * 1024;
    const __bf16* b2r = W + (long)(n0 + 32 + l16) * 1024;
    const __bf16* b3r = W + (long)(n0 + 48 + l16) * 1024;

    v8f acc[4] = {zero8(), zero8(), zero8(), zero8()};
    v16bf a  = load_a_bf16(arow, 0, half);
    v16bf b0 = load_b_bf16(b0r, 0, half);
    v16bf b1 = load_b_bf16(b1r, 0, half);
    v16bf b2 = load_b_bf16(b2r, 0, half);
    v16bf b3 = load_b_bf16(b3r, 0, half);
    for (int kb = 0; kb < 1024; kb += 32) {
        const int kn = (kb + 32) & 1023;          // branchless wrap on last iter
        v16bf an  = load_a_bf16(arow, kn, half);
        v16bf bn0 = load_b_bf16(b0r, kn, half);
        v16bf bn1 = load_b_bf16(b1r, kn, half);
        v16bf bn2 = load_b_bf16(b2r, kn, half);
        v16bf bn3 = load_b_bf16(b3r, kn, half);
        acc[0] = wmma_bf16(a, b0, acc[0]);
        acc[1] = wmma_bf16(a, b1, acc[1]);
        acc[2] = wmma_bf16(a, b2, acc[2]);
        acc[3] = wmma_bf16(a, b3, acc[3]);
        a = an; b0 = bn0; b1 = bn1; b2 = bn2; b3 = bn3;
    }

    for (int j = 0; j < 4; ++j) {
        const int n  = n0 + j * 16 + l16;
        const float bv = bias[n];
        const int h = n >> 6, dh = n & 63;
        for (int r = 0; r < 8; ++r) {
            const int t = m0 + r + 8 * half;   // global token
            const int b = t >> 11, s = t & 2047;
            const float val = acc[j][r] + bv;
            long idx;
            if (mode == 2) idx = ((long)((b * 16 + h) * 64 + dh)) * 2048 + s;
            else           idx = ((long)((b * 16 + h) * 2048 + s)) * 64 + dh;
            out[idx] = (__bf16)val;
        }
    }
}

// ---------------------------------------------------------------------------
// Fused attention: one workgroup (8 waves) per (bh, 16-query strip).
//  Phase 1: scores strip [16 x 2048] -> LDS (f32), WMMA QK^T * 0.125
//  Phase 2: softmax over LDS rows; write normalized probs to d_out (only
//           global traffic for attn)
//  Phase 3: AV from LDS probs (bf16-converted on the fly) x Vt; cross-wave
//           K-split reduced with ds_add_f32; context -> AO bf16
// Dynamic LDS: 16*2048 f32 scores + 16*64 ctx + reduction scratch (~136 KB)
// ---------------------------------------------------------------------------
extern __shared__ char smem_raw[];

__global__ __launch_bounds__(256) void attn_fused_kernel(const __bf16* __restrict__ Qh,
                                                         const __bf16* __restrict__ Kh,
                                                         const __bf16* __restrict__ Vt,
                                                         float* __restrict__ attn,
                                                         __bf16* __restrict__ AO) {
    float* sc   = (float*)smem_raw;      // [16][2048]
    float* ctx  = sc + 16 * 2048;        // [16][64]
    float* redA = ctx + 16 * 64;         // [256]
    float* rmax = redA + 256;            // [16]
    float* rsum = rmax + 16;             // [16]

    const int tid  = threadIdx.x;
    const int lane = tid & 31;
    const int w    = tid >> 5;           // wave 0..7
    const int half = lane >> 4, l16 = lane & 15;

    const int qtile = blockIdx.x & 127;
    const int bh    = blockIdx.x >> 7;   // 0..31
    const int q0    = qtile * 16;
    const int b     = bh >> 4, h = bh & 15;

    // zero the context accumulator
    for (int i = 0; i < 4; ++i) ctx[tid * 4 + i] = 0.f;

    // ---------------- Phase 1: scores -> LDS ----------------
    const __bf16* qrow  = Qh + ((long)bh * 2048 + q0 + l16) * 64;
    const __bf16* kbase = Kh + (long)bh * 2048 * 64;
    const v16bf a0 = load_a_bf16(qrow, 0, half);
    const v16bf a1 = load_a_bf16(qrow, 32, half);

    for (int i = 0; i < 4; ++i) {                 // 4 k-strips of 64 per wave
        const int k0 = (w * 4 + i) * 64;
        v8f acc[4] = {zero8(), zero8(), zero8(), zero8()};
        for (int j = 0; j < 4; ++j) {
            const __bf16* krow = kbase + (long)(k0 + j * 16 + l16) * 64;
            acc[j] = wmma_bf16(a0, load_b_bf16(krow, 0, half), acc[j]);
            acc[j] = wmma_bf16(a1, load_b_bf16(krow, 32, half), acc[j]);
        }
        for (int j = 0; j < 4; ++j) {
            const int kk = k0 + j * 16 + l16;
            for (int r = 0; r < 8; ++r)
                sc[(r + 8 * half) * 2048 + kk] = acc[j][r] * 0.125f;
        }
    }
    __syncthreads();

    // ---------------- Phase 2: softmax in LDS + write attn ----------------
    {
        const int row = tid >> 4;        // 16 threads per row
        const int sub = tid & 15;        // each handles 128 contiguous elems
        float* rp = sc + row * 2048 + sub * 128;

        float lmax = -3.402823466e38f;
        for (int i = 0; i < 128; i += 4) {
            v4f v = *(const v4f*)(rp + i);
            lmax = fmaxf(lmax, fmaxf(fmaxf(v[0], v[1]), fmaxf(v[2], v[3])));
        }
        redA[tid] = lmax; __syncthreads();
        if (sub == 0) {
            float m = redA[row * 16];
            for (int k = 1; k < 16; ++k) m = fmaxf(m, redA[row * 16 + k]);
            rmax[row] = m;
        }
        __syncthreads();
        const float m = rmax[row];

        float lsum = 0.f;
        for (int i = 0; i < 128; i += 4) {
            v4f v = *(const v4f*)(rp + i);
            v4f e;
            e[0] = __expf(v[0] - m); e[1] = __expf(v[1] - m);
            e[2] = __expf(v[2] - m); e[3] = __expf(v[3] - m);
            *(v4f*)(rp + i) = e;
            lsum += e[0] + e[1] + e[2] + e[3];
        }
        redA[tid] = lsum; __syncthreads();
        if (sub == 0) {
            float s = redA[row * 16];
            for (int k = 1; k < 16; ++k) s += redA[row * 16 + k];
            rsum[row] = s;
        }
        __syncthreads();
        const float inv = 1.f / rsum[row];

        float* gp = attn + ((long)bh * 2048 + q0 + row) * 2048 + sub * 128;
        for (int i = 0; i < 128; i += 4) {
            v4f v = *(const v4f*)(rp + i);
            v *= inv;
            *(v4f*)(rp + i) = v;          // keep normalized probs in LDS
            *(v4f*)(gp + i) = v;          // the only global attn traffic
        }
    }
    __syncthreads();

    // ---------------- Phase 3: AV from LDS probs ----------------
    {
        const __bf16* vb = Vt + (long)bh * 64 * 2048;
        const float* aLrow = sc + l16 * 2048;
        const int kblo = w * 256, kbhi = kblo + 256;   // K split across 8 waves

        v8f acc[4] = {zero8(), zero8(), zero8(), zero8()};
        for (int kb = kblo; kb < kbhi; kb += 32) {
            v16bf a = load_a_ldsf32(aLrow, kb, half);
            for (int j = 0; j < 4; ++j) {
                v16bf bb = load_b_bf16(vb + (long)(j * 16 + l16) * 2048, kb, half);
                acc[j] = wmma_bf16(a, bb, acc[j]);
            }
        }
        for (int j = 0; j < 4; ++j)
            for (int r = 0; r < 8; ++r)
                atomicAdd(&ctx[(r + 8 * half) * 64 + j * 16 + l16], acc[j][r]);
    }
    __syncthreads();

    // ---------------- Phase 4: store context as bf16 AO ----------------
    {
        const int idx = tid * 4;           // 1024 elems = 16 rows x 64 cols
        const int row = idx >> 6, col = idx & 63;
        __bf16* dst = AO + ((long)(b * 2048 + q0 + row)) * 1024 + h * 64 + col;
        for (int i = 0; i < 4; ++i) dst[i] = (__bf16)ctx[idx + i];
    }
}

// ---------------------------------------------------------------------------
// Output projection: out[4096,1024] f32 = AO x Wo^T + bo (double-buffered)
// ---------------------------------------------------------------------------
__global__ __launch_bounds__(256) void proj_out_kernel(const __bf16* __restrict__ AO,
                                                       const __bf16* __restrict__ Wo,
                                                       const float* __restrict__ bo,
                                                       float* __restrict__ out) {
    const int lane = threadIdx.x & 31;
    const int wid  = blockIdx.x * 8 + (threadIdx.x >> 5);
    const int strip = wid & 15;
    const int mtile = wid >> 4;
    const int m0 = mtile * 16, n0 = strip * 64;
    const int half = lane >> 4, l16 = lane & 15;

    const __bf16* arow = AO + (long)(m0 + l16) * 1024;
    const __bf16* b0r = Wo + (long)(n0 +  0 + l16) * 1024;
    const __bf16* b1r = Wo + (long)(n0 + 16 + l16) * 1024;
    const __bf16* b2r = Wo + (long)(n0 + 32 + l16) * 1024;
    const __bf16* b3r = Wo + (long)(n0 + 48 + l16) * 1024;

    v8f acc[4] = {zero8(), zero8(), zero8(), zero8()};
    v16bf a  = load_a_bf16(arow, 0, half);
    v16bf b0 = load_b_bf16(b0r, 0, half);
    v16bf b1 = load_b_bf16(b1r, 0, half);
    v16bf b2 = load_b_bf16(b2r, 0, half);
    v16bf b3 = load_b_bf16(b3r, 0, half);
    for (int kb = 0; kb < 1024; kb += 32) {
        const int kn = (kb + 32) & 1023;
        v16bf an  = load_a_bf16(arow, kn, half);
        v16bf bn0 = load_b_bf16(b0r, kn, half);
        v16bf bn1 = load_b_bf16(b1r, kn, half);
        v16bf bn2 = load_b_bf16(b2r, kn, half);
        v16bf bn3 = load_b_bf16(b3r, kn, half);
        acc[0] = wmma_bf16(a, b0, acc[0]);
        acc[1] = wmma_bf16(a, b1, acc[1]);
        acc[2] = wmma_bf16(a, b2, acc[2]);
        acc[3] = wmma_bf16(a, b3, acc[3]);
        a = an; b0 = bn0; b1 = bn1; b2 = bn2; b3 = bn3;
    }
    for (int j = 0; j < 4; ++j) {
        const int n = n0 + j * 16 + l16;
        const float bv = bo[n];
        for (int r = 0; r < 8; ++r) {
            const int t = m0 + r + 8 * half;
            out[(long)t * 1024 + n] = acc[j][r] + bv;
        }
    }
}

// ---------------------------------------------------------------------------
extern "C" void kernel_launch(void* const* d_in, const int* in_sizes, int n_in,
                              void* d_out, int out_size, void* d_ws, size_t ws_size,
                              hipStream_t stream) {
    const float* query = (const float*)d_in[0];
    const float* Wq = (const float*)d_in[1]; const float* bq = (const float*)d_in[2];
    const float* Wk = (const float*)d_in[3]; const float* bk = (const float*)d_in[4];
    const float* Wv = (const float*)d_in[5]; const float* bv = (const float*)d_in[6];
    const float* Wo = (const float*)d_in[7]; const float* bo = (const float*)d_in[8];

    char* ws = (char*)d_ws;
    __bf16* Xbf = (__bf16*)ws;                 ws += (size_t)4096 * 1024 * 2;  // 8 MB
    __bf16* Wqb = (__bf16*)ws;                 ws += (size_t)1024 * 1024 * 2;  // 2 MB
    __bf16* Wkb = (__bf16*)ws;                 ws += (size_t)1024 * 1024 * 2;
    __bf16* Wvb = (__bf16*)ws;                 ws += (size_t)1024 * 1024 * 2;
    __bf16* Wob = (__bf16*)ws;                 ws += (size_t)1024 * 1024 * 2;
    __bf16* Qh  = (__bf16*)ws;                 ws += (size_t)4096 * 1024 * 2;  // [b,h,s,dh]
    __bf16* Kh  = (__bf16*)ws;                 ws += (size_t)4096 * 1024 * 2;  // [b,h,s,dh]
    __bf16* Vt  = (__bf16*)ws;                 ws += (size_t)4096 * 1024 * 2;  // [b,h,dh,s]
    __bf16* AO  = (__bf16*)ws;                 ws += (size_t)4096 * 1024 * 2;  // [token,1024]

    float* out  = (float*)d_out;
    float* attn = out + (size_t)2 * 2048 * 1024;

    // 1) casts to bf16
    cast_f32_bf16<<<(4096 * 1024) / 256, 256, 0, stream>>>(query, Xbf, 4096 * 1024);
    cast_f32_bf16<<<(1024 * 1024) / 256, 256, 0, stream>>>(Wq, Wqb, 1024 * 1024);
    cast_f32_bf16<<<(1024 * 1024) / 256, 256, 0, stream>>>(Wk, Wkb, 1024 * 1024);
    cast_f32_bf16<<<(1024 * 1024) / 256, 256, 0, stream>>>(Wv, Wvb, 1024 * 1024);
    cast_f32_bf16<<<(1024 * 1024) / 256, 256, 0, stream>>>(Wo, Wob, 1024 * 1024);

    // 2) Q/K/V projections
    proj_qkv_kernel<<<512, 256, 0, stream>>>(Xbf, Wqb, bq, Qh, 0);
    proj_qkv_kernel<<<512, 256, 0, stream>>>(Xbf, Wkb, bk, Kh, 1);
    proj_qkv_kernel<<<512, 256, 0, stream>>>(Xbf, Wvb, bv, Vt, 2);

    // 3) fused scores + softmax + AV; LDS = scores strip + ctx + reductions
    const size_t smem_bytes = (16 * 2048 + 16 * 64 + 256 + 16 + 16) * sizeof(float);
    attn_fused_kernel<<<4096, 256, smem_bytes, stream>>>(Qh, Kh, Vt, attn, AO);

    // 4) output projection -> f32 out
    proj_out_kernel<<<512, 256, 0, stream>>>(AO, Wob, bo, out);
}
